// AttentionMechanism_25598005084434
// MI455X (gfx1250) — compile-verified, matrix operand order checked by reading, not run on hardware
//
#include <hip/hip_runtime.h>

// ---------------------------------------------------------------------------
// Problem dims
#define Bdim 8
#define Sdim 4096
#define Hdim 1024
#define MROWS 32768          // B*S rows
#define ROWS 64              // M rows per workgroup in the fused GEMM
#define NT_PER_WAVE 8        // 64 N-tiles of 16 cols, split over 8 waves
#define LSTRIDE 1040         // padded LDS row stride (elements): 520 dwords,
                             // 520 % 64 == 8 -> conflict-free b128 reads

// Plain (trivially-copyable) vector types for fragments / wide loads
typedef __attribute__((ext_vector_type(16))) __bf16        v16bf;
typedef __attribute__((ext_vector_type(8)))  float         v8f;
typedef __attribute__((ext_vector_type(4)))  unsigned int  u32x4;
typedef __attribute__((ext_vector_type(2)))  unsigned int  u32x2;
typedef __attribute__((ext_vector_type(4)))  float         f32x4;

union BF16Frag { u32x4 u[2]; v16bf v; };

__device__ __forceinline__ unsigned short f2bf(float f) {
  unsigned int u = __builtin_bit_cast(unsigned int, f);
  u += 0x7FFFu + ((u >> 16) & 1u);          // round-to-nearest-even
  return (unsigned short)(u >> 16);
}

// ---------------------------------------------------------------------------
// Kernel 1: convert W1/W2 to bf16 in workspace; zero vsum accumulator.
__global__ void __launch_bounds__(256) convert_init_kernel(
    const float* __restrict__ w1, const float* __restrict__ w2,
    unsigned short* __restrict__ w1bf, unsigned short* __restrict__ w2bf,
    float* __restrict__ vsum)
{
  const int i = blockIdx.x * 256 + threadIdx.x;
  if (i < 1048576) {
    w1bf[i] = f2bf(w1[i]);
  } else if (i < 2097152) {
    w2bf[i - 1048576] = f2bf(w2[i - 1048576]);
  } else if (i < 2097152 + Bdim * Hdim) {
    vsum[i - 2097152] = 0.0f;
  }
}

// ---------------------------------------------------------------------------
// Kernel 2: fused  s[row] = tanh(query@W1^T + key@W2^T + b1 + b2) . V_w + V_b
// One block = 64 rows; query/key staged in LDS as bf16 with padded stride
// (bank-conflict-free ds_load_b128).  bf16 WMMA 16x16x32; B fragment
// prefetched one K-step ahead; K loop unroll capped at 2 so the A-fragment
// LDS loads stay inside the loop (prevents hoist-and-spill to scratch).
__global__ void __launch_bounds__(256) score_gemm_kernel(
    const float* __restrict__ q, const float* __restrict__ k,
    const unsigned short* __restrict__ w1bf, const unsigned short* __restrict__ w2bf,
    const float* __restrict__ b1, const float* __restrict__ b2,
    const float* __restrict__ vw, const float* __restrict__ vb,
    float* __restrict__ sout)
{
  extern __shared__ unsigned short smem[];
  unsigned short* shq  = smem;                         // [ROWS][LSTRIDE] bf16
  unsigned short* shk  = smem + ROWS * LSTRIDE;        // [ROWS][LSTRIDE] bf16
  float*          srow = (float*)(smem + 2 * ROWS * LSTRIDE);  // [ROWS]

  const int  tid     = threadIdx.x;
  const long rowbase = (long)blockIdx.x * ROWS;
  const float* qg = q + rowbase * Hdim;
  const float* kg = k + rowbase * Hdim;

  // Stage 64 x 1024 of query and key into LDS, converting f32 -> bf16.
  for (int i = tid * 4; i < ROWS * Hdim; i += 256 * 4) {
    const int r = i >> 10;
    const int c = i & 1023;
    const int o = r * LSTRIDE + c;
    f32x4 vq = *(const f32x4*)(qg + i);
    f32x4 vk = *(const f32x4*)(kg + i);
    u32x2 pq, pk;
    pq.x = (unsigned)f2bf(vq.x) | ((unsigned)f2bf(vq.y) << 16);
    pq.y = (unsigned)f2bf(vq.z) | ((unsigned)f2bf(vq.w) << 16);
    pk.x = (unsigned)f2bf(vk.x) | ((unsigned)f2bf(vk.y) << 16);
    pk.y = (unsigned)f2bf(vk.z) | ((unsigned)f2bf(vk.w) << 16);
    *(u32x2*)(shq + o) = pq;
    *(u32x2*)(shk + o) = pk;
  }
  if (tid < ROWS) srow[tid] = 0.0f;
  __syncthreads();

  const int wave = tid >> 5;
  const int lane = tid & 31;
  const int ln   = lane & 15;          // N (or M) index within tile
  const int koff = (lane >> 4) << 3;   // 0 or 8 : K-half select per ISA layout

  float part[4][8];
  #pragma unroll
  for (int m = 0; m < 4; ++m)
    #pragma unroll
    for (int r = 0; r < 8; ++r) part[m][r] = 0.0f;

  for (int nt = 0; nt < NT_PER_WAVE; ++nt) {
    const int jbase = (wave * NT_PER_WAVE + nt) * 16;
    const int j     = jbase + ln;

    v8f acc[4];
    #pragma unroll
    for (int m = 0; m < 4; ++m)
      acc[m] = (v8f){0.f, 0.f, 0.f, 0.f, 0.f, 0.f, 0.f, 0.f};

    const unsigned short* wrow1 = w1bf + (size_t)j * Hdim + koff;
    const unsigned short* wrow2 = w2bf + (size_t)j * Hdim + koff;

    // ---- Phase 1: query @ W1^T (B fragment pipelined one K-step ahead) ----
    BF16Frag fb;
    { const u32x4* pb = (const u32x4*)wrow1; fb.u[0] = pb[0]; fb.u[1] = pb[2]; }
    #pragma unroll 2
    for (int kc = 0; kc < Hdim; kc += 32) {
      const int kn = (kc + 32 < Hdim) ? kc + 32 : kc;   // clamped prefetch
      BF16Frag fbn;
      { const u32x4* pb = (const u32x4*)(wrow1 + kn);
        fbn.u[0] = pb[0]; fbn.u[1] = pb[2]; }
      BF16Frag fa[4];
      #pragma unroll
      for (int m = 0; m < 4; ++m) {
        const u32x4* pa = (const u32x4*)(shq + (m * 16 + ln) * LSTRIDE + kc + koff);
        fa[m].u[0] = pa[0]; fa[m].u[1] = pa[2];
      }
      #pragma unroll
      for (int m = 0; m < 4; ++m)
        acc[m] = __builtin_amdgcn_wmma_f32_16x16x32_bf16(
            false, fa[m].v, false, fb.v, (short)0, acc[m], false, false);
      fb = fbn;
    }
    // ---- Phase 2: key @ W2^T (accumulates into same tile before tanh) ----
    { const u32x4* pb = (const u32x4*)wrow2; fb.u[0] = pb[0]; fb.u[1] = pb[2]; }
    #pragma unroll 2
    for (int kc = 0; kc < Hdim; kc += 32) {
      const int kn = (kc + 32 < Hdim) ? kc + 32 : kc;
      BF16Frag fbn;
      { const u32x4* pb = (const u32x4*)(wrow2 + kn);
        fbn.u[0] = pb[0]; fbn.u[1] = pb[2]; }
      BF16Frag fa[4];
      #pragma unroll
      for (int m = 0; m < 4; ++m) {
        const u32x4* pa = (const u32x4*)(shk + (m * 16 + ln) * LSTRIDE + kc + koff);
        fa[m].u[0] = pa[0]; fa[m].u[1] = pa[2];
      }
      #pragma unroll
      for (int m = 0; m < 4; ++m)
        acc[m] = __builtin_amdgcn_wmma_f32_16x16x32_bf16(
            false, fa[m].v, false, fb.v, (short)0, acc[m], false, false);
      fb = fbn;
    }

    // Epilogue: tanh(h + bias) * V_w[j], accumulated per (row) in registers.
    const float vwj  = vw[j];
    const float bias = b1[j] + b2[j];
    #pragma unroll
    for (int m = 0; m < 4; ++m)
      #pragma unroll
      for (int r = 0; r < 8; ++r)
        part[m][r] += tanhf(acc[m][r] + bias) * vwj;
  }

  // Reduce across the 16 N-lanes of each half-wave, then one LDS atomic/row.
  #pragma unroll
  for (int m = 0; m < 4; ++m) {
    #pragma unroll
    for (int r = 0; r < 8; ++r) {
      float v = part[m][r];
      v += __shfl_xor(v, 1);
      v += __shfl_xor(v, 2);
      v += __shfl_xor(v, 4);
      v += __shfl_xor(v, 8);
      if (ln == 0) atomicAdd(&srow[m * 16 + r + koff], v);
    }
  }
  __syncthreads();
  if (tid < ROWS) sout[rowbase + tid] = srow[tid] + vb[0];
}

// ---------------------------------------------------------------------------
// Kernel 3: vsum[b][h] = sum_s value[b,s,h]  (chunked S, global f32 atomics)
__global__ void __launch_bounds__(256) vsum_kernel(
    const float* __restrict__ value, float* __restrict__ vsum)
{
  const int g = blockIdx.x * 256 + threadIdx.x;   // 65536 threads
  const int c = g >> 13;                          // S-chunk 0..7 (512 each)
  const int r = g & 8191;                         // b*1024 + h
  const int b = r >> 10;
  const int h = r & 1023;
  const float* p = value + ((size_t)(b << 12) + (size_t)(c << 9)) * Hdim + h;
  float acc = 0.0f;
  for (int s = 0; s < 512; ++s) acc += p[(size_t)s << 10];
  atomicAdd(&vsum[r], acc);
}

// ---------------------------------------------------------------------------
// Kernel 4: softmax over S per batch. One block (1024 thr) per batch.
__global__ void __launch_bounds__(1024) softmax_kernel(
    const float* __restrict__ s, float* __restrict__ attn)
{
  __shared__ float red[1024];
  const int b = blockIdx.x, tid = threadIdx.x;
  const float* sb = s + (size_t)b * Sdim;

  float e[4];
  float m = -3.402823466e38f;
  #pragma unroll
  for (int j = 0; j < 4; ++j) { e[j] = sb[tid + j * 1024]; m = fmaxf(m, e[j]); }
  red[tid] = m; __syncthreads();
  for (int off = 512; off > 0; off >>= 1) {
    if (tid < off) red[tid] = fmaxf(red[tid], red[tid + off]);
    __syncthreads();
  }
  m = red[0]; __syncthreads();

  float sum = 0.0f;
  #pragma unroll
  for (int j = 0; j < 4; ++j) { e[j] = __expf(e[j] - m); sum += e[j]; }
  red[tid] = sum; __syncthreads();
  for (int off = 512; off > 0; off >>= 1) {
    if (tid < off) red[tid] += red[tid + off];
    __syncthreads();
  }
  const float inv = 1.0f / red[0];
  #pragma unroll
  for (int j = 0; j < 4; ++j)
    attn[(size_t)b * Sdim + tid + j * 1024] = e[j] * inv;
}

// ---------------------------------------------------------------------------
// Kernel 5: context[b,s,h] = attn[b,s] * vsum[b,h]  (float4 streaming write)
__global__ void __launch_bounds__(256) context_kernel(
    const float* __restrict__ attn, const float* __restrict__ vsum,
    float* __restrict__ ctx)
{
  const size_t g   = (size_t)blockIdx.x * 256 + threadIdx.x;  // float4 index
  const int    row = (int)(g >> 8);        // (b*S + s)
  const int    b   = row >> 12;
  const int    h4  = (int)(g & 255);
  const float  a   = attn[row];
  f32x4 vs = ((const f32x4*)vsum)[(b << 8) + h4];
  f32x4 o;
  o.x = a * vs.x; o.y = a * vs.y; o.z = a * vs.z; o.w = a * vs.w;
  ((f32x4*)ctx)[g] = o;
}

// ---------------------------------------------------------------------------
extern "C" void kernel_launch(void* const* d_in, const int* in_sizes, int n_in,
                              void* d_out, int out_size, void* d_ws, size_t ws_size,
                              hipStream_t stream) {
  const float* query = (const float*)d_in[0];
  const float* key_  = (const float*)d_in[1];
  const float* value = (const float*)d_in[2];
  const float* W1_w  = (const float*)d_in[3];
  const float* W1_b  = (const float*)d_in[4];
  const float* W2_w  = (const float*)d_in[5];
  const float* W2_b  = (const float*)d_in[6];
  const float* V_w   = (const float*)d_in[7];
  const float* V_b   = (const float*)d_in[8];

  char* ws = (char*)d_ws;
  unsigned short* w1bf = (unsigned short*)ws;                       // 2 MB
  unsigned short* w2bf = (unsigned short*)(ws + (2u << 20));        // 2 MB
  float* sbuf = (float*)(ws + (4u << 20));                          // 128 KB
  float* vsum = (float*)(ws + (4u << 20) + (1u << 17));             // 32 KB

  float* ctx  = (float*)d_out;                                      // B*S*H
  float* attn = ctx + (size_t)Bdim * Sdim * Hdim;                   // B*S

  // 1) weight convert + vsum zero   (2*1M + 8192 elems -> 8224 blocks exact)
  convert_init_kernel<<<8224, 256, 0, stream>>>(W1_w, W2_w, w1bf, w2bf, vsum);

  // 2) fused double-GEMM + tanh + V-dot  (512 blocks, ~260 KB dynamic LDS)
  const size_t lds_bytes =
      2u * ROWS * LSTRIDE * sizeof(unsigned short) + ROWS * sizeof(float);
  score_gemm_kernel<<<MROWS / ROWS, 256, lds_bytes, stream>>>(
      query, key_, w1bf, w2bf, W1_b, W2_b, V_w, V_b, sbuf);

  // 3) vsum reduction over S
  vsum_kernel<<<256, 256, 0, stream>>>(value, vsum);

  // 4) softmax over S per batch -> attn output region
  softmax_kernel<<<Bdim, 1024, 0, stream>>>(sbuf, attn);

  // 5) context = attn * vsum broadcast
  context_kernel<<<(Bdim * Sdim * Hdim / 4) / 256, 256, 0, stream>>>(attn, vsum, ctx);
}
